// TemporalCrossAttention_91319594647642
// MI455X (gfx1250) — matrix operand order repaired and checked by reading, not compiled, not run on hardware
//
#include <hip/hip_runtime.h>
#include <hip/hip_bf16.h>

// ---------------- problem constants ----------------
constexpr int Cc  = 256;                 // channels
constexpr int NH  = 8;
constexpr int DH  = 32;                  // head dim = one WMMA K step
constexpr int Vv  = 6, Tt = 3, HIN = 32, WIN = 32;
constexpr int LQ  = 32 * 32;             // 1024 queries
constexpr int LK  = Vv * Tt * HIN * WIN; // 18432 keys
constexpr int KC  = 128;                 // keys per flash iteration
constexpr float LN_EPS = 1e-5f;
// (1/sqrt(dh)) * log2(e): fold attention scale + base-2 exp into Q
constexpr float QSCALE = 0.2550695314903178f;

typedef __attribute__((ext_vector_type(16))) __bf16 v16bf;
typedef __attribute__((ext_vector_type(8)))  __bf16 v8bf;
typedef __attribute__((ext_vector_type(8)))  float  v8f;

__device__ __forceinline__ v8f wmma_bf16(v16bf a, v16bf b, v8f c) {
  return __builtin_amdgcn_wmma_f32_16x16x32_bf16(false, a, false, b,
                                                 (short)0, c, false, false);
}
// 16 contiguous bf16 -> operand register pair
__device__ __forceinline__ v16bf ld16(const __bf16* p) {
  v8bf lo = *(const v8bf*)p;
  v8bf hi = *(const v8bf*)(p + 8);
  return __builtin_shufflevector(lo, hi, 0,1,2,3,4,5,6,7,8,9,10,11,12,13,14,15);
}
// two 8-elt chunks (callers pass p1 = p0+8: contiguous K = {kb..kb+15})
__device__ __forceinline__ v16bf ld8x2(const __bf16* p0, const __bf16* p1) {
  v8bf lo = *(const v8bf*)p0;
  v8bf hi = *(const v8bf*)p1;
  return __builtin_shufflevector(lo, hi, 0,1,2,3,4,5,6,7,8,9,10,11,12,13,14,15);
}

// ---- 16-lane butterfly reductions on the VALU via DPP16 ----
// xor1 = quad_perm[1,0,3,2]=0xB1, xor2 = quad_perm[2,3,0,1]=0x4E,
// then row_half_mirror(0x141) and row_mirror(0x140) complete the 16-lane tree.
__device__ __forceinline__ float dpp_qmax(float x, const int ctrl_unused) { return x; }

template <int CTRL>
__device__ __forceinline__ float dpp_bcast_step_max(float x) {
  union { float f; int i; } a, b;
  a.f = x;
  b.i = __builtin_amdgcn_update_dpp(a.i, a.i, CTRL, 0xF, 0xF, true);
  return fmaxf(a.f, b.f);
}
template <int CTRL>
__device__ __forceinline__ float dpp_bcast_step_add(float x) {
  union { float f; int i; } a, b;
  a.f = x;
  b.i = __builtin_amdgcn_update_dpp(a.i, a.i, CTRL, 0xF, 0xF, true);
  return a.f + b.f;
}
__device__ __forceinline__ float red16_max(float x) {
  x = dpp_bcast_step_max<0xB1>(x);   // xor 1
  x = dpp_bcast_step_max<0x4E>(x);   // xor 2
  x = dpp_bcast_step_max<0x141>(x);  // row_half_mirror (~xor 4)
  x = dpp_bcast_step_max<0x140>(x);  // row_mirror      (~xor 8)
  return x;
}
__device__ __forceinline__ float red16_add(float x) {
  x = dpp_bcast_step_add<0xB1>(x);
  x = dpp_bcast_step_add<0x4E>(x);
  x = dpp_bcast_step_add<0x141>(x);
  x = dpp_bcast_step_add<0x140>(x);
  return x;
}

// ---------------- 1) query vector + projection (tiny matvec) -------------
__global__ void k_build_q(const float* __restrict__ qe,
                          const float* __restrict__ cpe,
                          const float* __restrict__ spe,
                          const float* __restrict__ qw,
                          const float* __restrict__ qb,
                          const int*   __restrict__ qci,
                          float* __restrict__ qp) {
  const int m   = threadIdx.x;           // 0..255
  const int cam = qci[0];
  float acc = qb[m];
  #pragma unroll 8
  for (int c = 0; c < Cc; ++c) {
    float qv = qe[c] + cpe[cam * Cc + c] + spe[c];
    acc = fmaf(qv, qw[m * Cc + c], acc);
  }
  qp[m] = acc;
}

// broadcast + fold softmax scale (base-2) into Q
__global__ void k_bcast_q(const float* __restrict__ qp, __bf16* __restrict__ Qb) {
  const int i = blockIdx.x * 256 + threadIdx.x;     // LQ*C threads
  Qb[i] = (__bf16)(qp[i & (Cc - 1)] * QSCALE);
}

// ---------------- 2) weight f32 -> bf16 ----------------
__global__ void k_cvt_w(const float* __restrict__ w, __bf16* __restrict__ o) {
  const int i = blockIdx.x * 256 + threadIdx.x;
  o[i] = (__bf16)w[i];
}

// ---------------- 3) build kv = hist + embeds, transpose to [n][C] bf16 ----
__global__ __launch_bounds__(256)
void k_build_kv(const float* __restrict__ hist,
                const float* __restrict__ tpe,
                const float* __restrict__ cpe,
                __bf16* __restrict__ kv) {
  __shared__ float tile[32][33];
  const int vt = blockIdx.z;             // 0..17 (v*T + t)
  const int v  = vt / Tt, t = vt % Tt;
  const int c0 = blockIdx.y * 32;
  const int s0 = blockIdx.x * 32;        // spatial y*32+x
  const int tx = threadIdx.x, ty = threadIdx.y;   // (32,8)
  #pragma unroll
  for (int i = 0; i < 4; ++i) {
    const int c = c0 + ty + i * 8;
    float x = hist[(size_t)(vt * Cc + c) * 1024 + s0 + tx]
            + tpe[t * Cc + c] + cpe[v * Cc + c];
    tile[ty + i * 8][tx] = x;
  }
  __syncthreads();
  #pragma unroll
  for (int i = 0; i < 4; ++i) {
    const int s = s0 + ty + i * 8;
    kv[(size_t)(vt * 1024 + s) * Cc + c0 + tx] = (__bf16)tile[tx][ty + i * 8];
  }
}

// ---------------- 4) WMMA GEMM: Out[n,m] = A[n,:]·W[m,:] + bias[m] --------
// A bf16 [L x C], W bf16 [C x C] (torch Linear weight, row = out ch).
// Each wave: 64x64 tile (4x4 of 16x16), block = 4 waves -> full 256 cols.
// OUT_MODE: 0 = bf16 [n*C+m], 1 = bf16 transposed [m*L+n], 2 = f32 [n*C+m]
template <int OUT_MODE>
__global__ __launch_bounds__(128)
void k_gemm(const __bf16* __restrict__ A, const __bf16* __restrict__ W,
            const float* __restrict__ bias, void* __restrict__ Out, int L) {
  const int lane = threadIdx.x & 31;
  const int wave = threadIdx.x >> 5;
  const int n0   = blockIdx.x * 64;
  const int m0   = wave * 64;
  const int arow = lane & 15;
  const int akb  = (lane < 16) ? 0 : 16;  // A lane K-base (contiguous, matches B)
  const int bcol = lane & 15;
  const int bkb  = (lane < 16) ? 0 : 16;  // B lane K-base

  v8f acc[4][4];
  #pragma unroll
  for (int i = 0; i < 4; ++i)
    #pragma unroll
    for (int j = 0; j < 4; ++j) { v8f z = {}; acc[i][j] = z; }

  for (int k0 = 0; k0 < Cc; k0 += 32) {
    v16bf a[4], b[4];
    #pragma unroll
    for (int i = 0; i < 4; ++i) {
      const __bf16* pa = A + (size_t)(n0 + i * 16 + arow) * Cc + k0 + akb;
      a[i] = ld8x2(pa, pa + 8);
    }
    #pragma unroll
    for (int j = 0; j < 4; ++j) {
      const __bf16* pb = W + (size_t)(m0 + j * 16 + bcol) * Cc + k0 + bkb;
      b[j] = ld16(pb);
    }
    #pragma unroll
    for (int i = 0; i < 4; ++i)
      #pragma unroll
      for (int j = 0; j < 4; ++j)
        acc[i][j] = wmma_bf16(a[i], b[j], acc[i][j]);
  }

  const int half8 = (lane < 16) ? 0 : 8;
  #pragma unroll
  for (int j = 0; j < 4; ++j) {
    const int   m  = m0 + j * 16 + bcol;
    const float bv = bias[m];
    #pragma unroll
    for (int i = 0; i < 4; ++i) {
      #pragma unroll
      for (int g = 0; g < 8; ++g) {
        const int n = n0 + i * 16 + g + half8;
        const float val = acc[i][j][g] + bv;
        if constexpr (OUT_MODE == 0)
          ((__bf16*)Out)[(size_t)n * Cc + m] = (__bf16)val;
        else if constexpr (OUT_MODE == 1)
          ((__bf16*)Out)[(size_t)m * L + n] = (__bf16)val;   // V transposed
        else
          ((float*)Out)[(size_t)n * Cc + m] = val;
      }
    }
  }
}

// ---------------- 5) flash attention, 128-key chunks ----------------------
// One 16-query tile of one head per wave; Q pre-scaled by log2(e)/sqrt(dh),
// softmax done in base 2.  Per chunk: 8 S-WMMAs, in-lane tile max/sum +
// single DPP16 butterfly per row, 8 PV-WMMAs.
__global__ __launch_bounds__(128)
void k_flash(const __bf16* __restrict__ Q, const __bf16* __restrict__ K,
             const __bf16* __restrict__ Vt, __bf16* __restrict__ Aout) {
  __shared__ __bf16 pbuf[4][16][KC];     // P staging, 4KB/wave
  const int lane = threadIdx.x & 31;
  const int wave = threadIdx.x >> 5;
  const int wid  = blockIdx.x * 4 + wave;   // 0..511
  const int head = wid >> 6;
  const int q0   = (wid & 63) * 16;
  const int arow = lane & 15;
  const int akb  = (lane < 16) ? 0 : 16;
  const int bcol = lane & 15;
  const int bkb  = (lane < 16) ? 0 : 16;

  const __bf16* pq = Q + (size_t)(q0 + arow) * Cc + head * DH + akb;
  const v16bf aq = ld8x2(pq, pq + 8);

  v8f o0 = {}, o1 = {};
  float mrow[8], lrow[8];
  #pragma unroll
  for (int g = 0; g < 8; ++g) { mrow[g] = -1e30f; lrow[g] = 0.f; }

  __bf16* myp = &pbuf[wave][0][0];
  const int prow = ((lane < 16) ? 0 : 8);
  const int pcol = lane & 15;

  for (int kn0 = 0; kn0 < LK; kn0 += KC) {
    // ---- S = Q · K^T : 8 independent 16x16 score tiles ----
    v8f s[8];
    #pragma unroll
    for (int j = 0; j < 8; ++j) {
      const __bf16* pk = K + (size_t)(kn0 + j * 16 + bcol) * Cc + head * DH + bkb;
      v8f z = {};
      s[j] = wmma_bf16(aq, ld16(pk), z);
    }

    // ---- online softmax (base 2); stats layout == C/D accumulator rows ----
    #pragma unroll
    for (int g = 0; g < 8; ++g) {
      float mx = s[0][g];
      #pragma unroll
      for (int j = 1; j < 8; ++j) mx = fmaxf(mx, s[j][g]);   // in-lane, 8 tiles
      mx = red16_max(mx);                                    // one DPP butterfly
      const float mnew  = fmaxf(mrow[g], mx);
      const float alpha = exp2f(mrow[g] - mnew);
      mrow[g] = mnew;
      o0[g] *= alpha;
      o1[g] *= alpha;
      float rs = 0.f;
      #pragma unroll
      for (int j = 0; j < 8; ++j) {
        const float p = exp2f(s[j][g] - mnew);
        rs += p;
        myp[(g + prow) * KC + j * 16 + pcol] = (__bf16)p;    // D-layout -> LDS
      }
      rs = red16_add(rs);                                    // one DPP butterfly
      lrow[g] = lrow[g] * alpha + rs;
    }
    asm volatile("s_wait_dscnt 0" ::: "memory");             // LDS RAW fence

    // ---- O += P · V : 4 k-steps of 32 keys, 2 dh tiles ----
    #pragma unroll
    for (int kk = 0; kk < 4; ++kk) {
      const __bf16* pp = myp + arow * KC + kk * 32 + akb;
      const v16bf ap = ld8x2(pp, pp + 8);
      const __bf16* pv0 = Vt + (size_t)(head * DH + bcol) * LK + kn0 + kk * 32 + bkb;
      const __bf16* pv1 = Vt + (size_t)(head * DH + 16 + bcol) * LK + kn0 + kk * 32 + bkb;
      o0 = wmma_bf16(ap, ld16(pv0), o0);
      o1 = wmma_bf16(ap, ld16(pv1), o1);
    }
  }

  #pragma unroll
  for (int g = 0; g < 8; ++g) {
    const float inv = 1.0f / lrow[g];
    const int q = q0 + g + prow;
    Aout[(size_t)q * Cc + head * DH + pcol]      = (__bf16)(o0[g] * inv);
    Aout[(size_t)q * Cc + head * DH + 16 + pcol] = (__bf16)(o1[g] * inv);
  }
}

// ---------------- 6) LayerNorm + transpose to (C,H,W) ----------------
__global__ __launch_bounds__(256)
void k_ln(const float* __restrict__ X, const float* __restrict__ gam,
          const float* __restrict__ bet, float* __restrict__ out) {
  const int n = blockIdx.x;          // query position
  const int c = threadIdx.x;         // channel
  float x = X[(size_t)n * Cc + c];
  float s = x, s2 = x * x;
  #pragma unroll
  for (int off = 16; off > 0; off >>= 1) {
    s  += __shfl_xor(s,  off, 32);
    s2 += __shfl_xor(s2, off, 32);
  }
  __shared__ float ws1[8], ws2[8];
  const int wv = threadIdx.x >> 5, ln = threadIdx.x & 31;
  if (ln == 0) { ws1[wv] = s; ws2[wv] = s2; }
  __syncthreads();
  float ts = 0.f, ts2 = 0.f;
  #pragma unroll
  for (int i = 0; i < 8; ++i) { ts += ws1[i]; ts2 += ws2[i]; }
  const float mu  = ts * (1.0f / Cc);
  const float var = ts2 * (1.0f / Cc) - mu * mu;
  const float y = (x - mu) * rsqrtf(var + LN_EPS) * gam[c] + bet[c];
  out[(size_t)c * LQ + n] = y;       // (B,C,H,W)
}

// ---------------- host launcher ----------------
extern "C" void kernel_launch(void* const* d_in, const int* in_sizes, int n_in,
                              void* d_out, int out_size, void* d_ws, size_t ws_size,
                              hipStream_t stream) {
  (void)in_sizes; (void)n_in; (void)out_size; (void)ws_size;
  const float* hist = (const float*)d_in[0];
  const float* qe   = (const float*)d_in[1];
  const float* cpe  = (const float*)d_in[2];
  const float* tpe  = (const float*)d_in[3];
  const float* spe  = (const float*)d_in[4];
  const float* qw   = (const float*)d_in[5];
  const float* qb   = (const float*)d_in[6];
  const float* kw   = (const float*)d_in[7];
  const float* kb   = (const float*)d_in[8];
  const float* vw   = (const float*)d_in[9];
  const float* vb   = (const float*)d_in[10];
  const float* ow   = (const float*)d_in[11];
  const float* ob   = (const float*)d_in[12];
  const float* lng  = (const float*)d_in[13];
  const float* lnb  = (const float*)d_in[14];
  const int*   qci  = (const int*)d_in[15];

  char* ws = (char*)d_ws;
  size_t off = 0;
  auto take = [&](size_t bytes) {
    size_t cur = off;
    off += (bytes + 255) & ~(size_t)255;
    return cur;
  };
  float*  qp  = (float*) (ws + take((size_t)Cc * 4));
  __bf16* Qb  = (__bf16*)(ws + take((size_t)LQ * Cc * 2));
  __bf16* KVb = (__bf16*)(ws + take((size_t)LK * Cc * 2));
  __bf16* Kb  = (__bf16*)(ws + take((size_t)LK * Cc * 2));
  __bf16* Vtb = (__bf16*)(ws + take((size_t)LK * Cc * 2));
  __bf16* Ab  = (__bf16*)(ws + take((size_t)LQ * Cc * 2));
  float*  Ob  = (float*) (ws + take((size_t)LQ * Cc * 4));
  __bf16* Wk  = (__bf16*)(ws + take((size_t)Cc * Cc * 2));
  __bf16* Wv  = (__bf16*)(ws + take((size_t)Cc * Cc * 2));
  __bf16* Wo  = (__bf16*)(ws + take((size_t)Cc * Cc * 2));

  // query path
  k_build_q<<<1, 256, 0, stream>>>(qe, cpe, spe, qw, qb, qci, qp);
  k_bcast_q<<<(LQ * Cc) / 256, 256, 0, stream>>>(qp, Qb);
  // weights -> bf16
  k_cvt_w<<<(Cc * Cc) / 256, 256, 0, stream>>>(kw, Wk);
  k_cvt_w<<<(Cc * Cc) / 256, 256, 0, stream>>>(vw, Wv);
  k_cvt_w<<<(Cc * Cc) / 256, 256, 0, stream>>>(ow, Wo);
  // kv build (transpose + pos embeds)
  k_build_kv<<<dim3(32, 8, Vv * Tt), dim3(32, 8), 0, stream>>>(hist, tpe, cpe, KVb);
  // K / V projections (V stored transposed [dh][LK] for PV B-operand loads)
  k_gemm<0><<<LK / 64, 128, 0, stream>>>(KVb, Wk, kb, (void*)Kb, LK);
  k_gemm<1><<<LK / 64, 128, 0, stream>>>(KVb, Wv, vb, (void*)Vtb, LK);
  // flash attention: 512 waves, 4 waves/block
  k_flash<<<128, 128, 0, stream>>>(Qb, Kb, Vtb, Ab);
  // output projection + LayerNorm + transpose
  k_gemm<2><<<LQ / 64, 128, 0, stream>>>(Ab, Wo, ob, (void*)Ob, LQ);
  k_ln<<<LQ, 256, 0, stream>>>(Ob, lng, lnb, (float*)d_out);
}
